// LMLTransformer_14783277433189
// MI455X (gfx1250) — matrix operand order repaired and checked by reading, not compiled, not run on hardware
//
#include <hip/hip_runtime.h>

typedef _Float16 f16;
typedef __attribute__((ext_vector_type(16))) _Float16 v16h;
typedef __attribute__((ext_vector_type(8)))  float    v8f;

#define WMMA_F16(a, b, c) \
  __builtin_amdgcn_wmma_f32_16x16x32_f16(false, (a), false, (b), (short)0, (c), false, false)

union Frag {
  v16h h16;
  f16 h[16];
  uint4 q4[2];
};

// ---- Fragment loaders (CDNA5 ISA 7.12.2 layouts, wave32), b128 LDS accesses ----

// A 16x32 f16 fragment from row-major [rows][pitch] LDS (pitch=72 halves keeps 16B align
// and conflict-free banks: 36*r mod 64 distinct for 16 rows).
__device__ __forceinline__ v16h load_a32(const f16* p, int pitch, int row0, int k0, int lane) {
  Frag f;
  const int base = (row0 + (lane & 15)) * pitch + k0 + ((lane >> 4) << 3);
  f.q4[0] = *(const uint4*)(p + base);        // halves 0..7  : K = kb .. kb+7
  f.q4[1] = *(const uint4*)(p + base + 16);   // halves 8..15 : K = kb+16 .. kb+23
  return f.h16;
}

// A fragment with only K=0..15 valid (zero-padded to K=32)
__device__ __forceinline__ v16h load_a16(const f16* p, int pitch, int row0, int lane) {
  Frag f;
  const int base = (row0 + (lane & 15)) * pitch + ((lane >> 4) << 3);
  f.q4[0] = *(const uint4*)(p + base);
  f.q4[1] = make_uint4(0u, 0u, 0u, 0u);
  return f.h16;
}

// B 32x16 fragment from TRANSPOSED storage [N][pitch(K)]: halves 0..7 K = kb..kb+7, halves 8..15 K = kb+16..kb+23.
__device__ __forceinline__ v16h load_bT32(const f16* pT, int pitch, int k0, int n0, int lane) {
  Frag f;
  const int base = (n0 + (lane & 15)) * pitch + k0 + ((lane >> 4) << 3);
  f.q4[0] = *(const uint4*)(pT + base);
  f.q4[1] = *(const uint4*)(pT + base + 16);
  return f.h16;
}

// B fragment for B = M^T, mat row-major [row][pitch], only 16 K-dims valid (rest zero).
__device__ __forceinline__ v16h load_bT16(const f16* mat, int pitch, int n0, int lane) {
  Frag f;
  {
    const f16* src = mat + (n0 + (lane & 15)) * pitch + ((lane >> 4) << 3);
    f.q4[0] = *(const uint4*)(src);
    f.q4[1] = make_uint4(0u, 0u, 0u, 0u);
  }
  return f.h16;
}

// prep: w_qkv (64x192) -> f16 transposed [192][64]; w_out (64x64) -> f16 transposed [64][64]
__global__ void lml_prep_weights(const float* __restrict__ wqkv,
                                 const float* __restrict__ wout,
                                 f16* __restrict__ ws) {
  int i = blockIdx.x * 256 + threadIdx.x;
  if (i < 12288) {
    int k = i / 192, n = i % 192;
    ws[n * 64 + k] = (f16)wqkv[i];
  } else if (i < 12288 + 4096) {
    int j = i - 12288;
    int k = j / 64, n = j % 64;
    ws[12288 + n * 64 + k] = (f16)wout[j];
  }
}

// Fused: qkv proj -> windowed SDPA (+LePE depthwise 3x3) -> out proj.
// One block per 8x8 window, 4 waves = 4 heads. Hardcoded H=W=512, DIM=64.
__global__ __launch_bounds__(128) void lml_fused(
    const float* __restrict__ xg, const f16* __restrict__ wsh,
    const float* __restrict__ bqkv, const float* __restrict__ wpe,
    const float* __restrict__ bpe,  const float* __restrict__ bout,
    float* __restrict__ outg)
{
  // LDS pool (halves, pitch 72): x 4608 | q 4608 | k 4608 | vT 4608 | wqT 13824 = 64512 B
  __shared__ alignas(16) f16 smem[32256];
  f16* lds_x   = smem;
  f16* lds_q   = smem + 4608;
  f16* lds_k   = smem + 9216;
  f16* lds_vT  = smem + 13824;   // [dim][token]
  f16* lds_wqT = smem + 18432;   // [n=192][k=64] pitch 72

  const int tid  = threadIdx.x;
  const int lane = tid & 31;
  const int wv   = tid >> 5;              // wave id == head id == M-block id
  const int hi8  = (lane >> 4) << 3;      // C-layout row offset, upper lane half
  const int lN   = lane & 15;

  const int win   = blockIdx.x;
  const int bb    = win >> 12;            // (512/8)^2 = 4096 windows per image
  const int wy    = (win >> 6) & 63;
  const int wx    = win & 63;
  const int row0g = wy << 3, col0g = wx << 3;

  // ---- Phase 1: stage x window (f32->f16) and w_qkv^T into LDS ----
  {
    const int token = tid >> 1;
    const int cb    = (tid & 1) << 5;
    const int sr = token >> 3, scc = token & 7;
    const int gtok = ((bb << 9) + row0g + sr) * 512 + (col0g + scc);
    const float4* src = (const float4*)(xg + (size_t)gtok * 64 + cb);
    f16* dst = lds_x + token * 72 + cb;
#pragma unroll
    for (int j = 0; j < 8; ++j) {
      float4 fv = src[j];
      union { f16 h[4]; uint2 u; } t4;
      t4.h[0] = (f16)fv.x; t4.h[1] = (f16)fv.y;
      t4.h[2] = (f16)fv.z; t4.h[3] = (f16)fv.w;
      *(uint2*)(dst + j * 4) = t4.u;
    }
    const uint4* s0 = (const uint4*)wsh;       // 192 rows x 8 uint4
    for (int i = tid; i < 1536; i += 128) {
      const int n = i >> 3, c = i & 7;
      *(uint4*)(lds_wqT + n * 72 + c * 8) = s0[i];
    }
  }
  __syncthreads();

  // ---- Phase 2: GEMM1 qkv[64x192] = x @ w_qkv + b; wave wv owns row-block mt=wv ----
  // Softmax scale 1/sqrt(16) folded into q epilogue.
  {
    const int mt = wv;
    const v16h a0 = load_a32(lds_x, 72, mt * 16, 0,  lane);
    const v16h a1 = load_a32(lds_x, 72, mt * 16, 32, lane);
#pragma unroll
    for (int nt = 0; nt < 12; ++nt) {
      v8f acc = {};
      acc = WMMA_F16(a0, load_bT32(lds_wqT, 72, 0,  nt * 16, lane), acc);
      acc = WMMA_F16(a1, load_bT32(lds_wqT, 72, 32, nt * 16, lane), acc);
      const float bias = bqkv[nt * 16 + lN];
#pragma unroll
      for (int r = 0; r < 8; ++r) {
        const int token = mt * 16 + r + hi8;
        if (nt < 4)       lds_q[token * 72 + nt * 16 + lN] = (f16)((acc[r] + bias) * 0.25f);
        else if (nt < 8)  lds_k[token * 72 + (nt - 4) * 16 + lN] = (f16)(acc[r] + bias);
        else              lds_vT[((nt - 8) * 16 + lN) * 72 + token] = (f16)(acc[r] + bias);
      }
    }
  }
  __syncthreads();

  // ---- Phase 3: attention head h = wv. Compute S^T = k @ q_scaled^T so the softmax
  // reduction axis (keys) lies inside each lane's registers (+ one lane^16 partner). ----
  const int h = wv;
  v16h bq[4];
#pragma unroll
  for (int mt = 0; mt < 4; ++mt) bq[mt] = load_bT16(lds_q + h * 16, 72, mt * 16, lane);
  v8f ST[4][4];  // [key tile nt][query tile mt]
#pragma unroll
  for (int nt = 0; nt < 4; ++nt) {
    const v16h ak = load_a16(lds_k + h * 16, 72, nt * 16, lane);
#pragma unroll
    for (int mt = 0; mt < 4; ++mt) {
      v8f z = {};
      ST[nt][mt] = WMMA_F16(ak, bq[mt], z);
    }
  }
  __syncthreads();  // q, k, x, wqT dead -> reuse as per-wave P buffers

  // Softmax over keys; write P row-major [query m][key n] for the O-GEMM A-fragments.
  f16* pbuf = (wv == 0) ? lds_x : (wv == 1) ? lds_q : (wv == 2) ? lds_k : lds_wqT;
#pragma unroll
  for (int mt = 0; mt < 4; ++mt) {
    float mx = ST[0][mt][0];
#pragma unroll
    for (int nt = 0; nt < 4; ++nt)
#pragma unroll
      for (int r = 0; r < 8; ++r) mx = fmaxf(mx, ST[nt][mt][r]);
    mx = fmaxf(mx, __shfl_xor(mx, 16, 32));
    float sum = 0.0f;
#pragma unroll
    for (int nt = 0; nt < 4; ++nt)
#pragma unroll
      for (int r = 0; r < 8; ++r) {
        const float e = __expf(ST[nt][mt][r] - mx);
        ST[nt][mt][r] = e;
        sum += e;
      }
    sum += __shfl_xor(sum, 16, 32);
    const float inv = 1.0f / sum;
    f16* pm = pbuf + (mt * 16 + lN) * 72 + hi8;   // row m = mt*16 + lane&15
#pragma unroll
    for (int nt = 0; nt < 4; ++nt)
#pragma unroll
      for (int r = 0; r < 8; ++r)
        pm[nt * 16 + r] = (f16)(ST[nt][mt][r] * inv);   // col n = nt*16 + r + hi8
  }
  __syncthreads();

  // ---- Phase 4: O = P @ v (B from vT), then LePE depthwise 3x3 ----
  const v16h bv0 = load_bT32(lds_vT, 72, 0,  h * 16, lane);
  const v16h bv1 = load_bT32(lds_vT, 72, 32, h * 16, lane);
  v8f o[4];
#pragma unroll
  for (int mt = 0; mt < 4; ++mt) {
    v8f acc = {};
    acc = WMMA_F16(load_a32(pbuf, 72, mt * 16, 0,  lane), bv0, acc);
    acc = WMMA_F16(load_a32(pbuf, 72, mt * 16, 32, lane), bv1, acc);
    o[mt] = acc;
  }
  // LePE: lane's 8 C-rows of tile mt are exactly window spatial row y0 = 2*mt + hi8/8,
  // columns tx = r (compile-time). Load 3 neighbor rows of this lane's fixed channel as
  // single b128 each; zero registers emulate the zero padding.
  {
    const int ch = h * 16 + lN;
    float w9[9];
#pragma unroll
    for (int j = 0; j < 9; ++j) w9[j] = wpe[ch * 9 + j];
    const float bp = bpe[ch];
    const f16* vrow = lds_vT + ch * 72;  // 64 tokens for this channel, 16B-aligned rows
    const int hb = hi8 >> 3;             // 0 or 1
#pragma unroll
    for (int mt = 0; mt < 4; ++mt) {
      const int y0 = 2 * mt + hb;
      union R { f16 h[8]; uint4 u; } rm, rc, rp;
      rc.u = *(const uint4*)(vrow + y0 * 8);
      if (y0 > 0) rm.u = *(const uint4*)(vrow + (y0 - 1) * 8);
      else        rm.u = make_uint4(0u, 0u, 0u, 0u);
      if (y0 < 7) rp.u = *(const uint4*)(vrow + (y0 + 1) * 8);
      else        rp.u = make_uint4(0u, 0u, 0u, 0u);
#pragma unroll
      for (int r = 0; r < 8; ++r) {
        float acc = bp;
        if (r > 0) {
          acc += w9[0] * (float)rm.h[r - 1];
          acc += w9[3] * (float)rc.h[r - 1];
          acc += w9[6] * (float)rp.h[r - 1];
        }
        acc += w9[1] * (float)rm.h[r];
        acc += w9[4] * (float)rc.h[r];
        acc += w9[7] * (float)rp.h[r];
        if (r < 7) {
          acc += w9[2] * (float)rm.h[r + 1];
          acc += w9[5] * (float)rc.h[r + 1];
          acc += w9[8] * (float)rp.h[r + 1];
        }
        o[mt][r] += acc;
      }
    }
  }
  __syncthreads();  // P and vT dead

  // ---- Phase 5: GEMM2 out = O @ w_out + b ----
  f16* lds_o   = lds_x;   // reuse, pitch 72
  f16* lds_woT = lds_q;   // reuse, [n=64][k=64] pitch 72 (4608 halves, fits q slot)
#pragma unroll
  for (int mt = 0; mt < 4; ++mt)
#pragma unroll
    for (int r = 0; r < 8; ++r)
      lds_o[(mt * 16 + r + hi8) * 72 + h * 16 + lN] = (f16)o[mt][r];
  {
    const uint4* s2 = (const uint4*)(wsh + 12288);  // 64 rows x 8 uint4
    for (int i = tid; i < 512; i += 128) {
      const int n = i >> 3, c = i & 7;
      *(uint4*)(lds_woT + n * 72 + c * 8) = s2[i];
    }
  }
  __syncthreads();

  {
    const int mt = wv;
    const v16h a0 = load_a32(lds_o, 72, mt * 16, 0,  lane);
    const v16h a1 = load_a32(lds_o, 72, mt * 16, 32, lane);
#pragma unroll
    for (int nt = 0; nt < 4; ++nt) {
      v8f acc = {};
      acc = WMMA_F16(a0, load_bT32(lds_woT, 72, 0,  nt * 16, lane), acc);
      acc = WMMA_F16(a1, load_bT32(lds_woT, 72, 32, nt * 16, lane), acc);
      const float bias = bout[nt * 16 + lN];
#pragma unroll
      for (int r = 0; r < 8; ++r) {
        const int token = mt * 16 + r + hi8;
        const int sr = token >> 3, scc = token & 7;
        const int gtok = ((bb << 9) + row0g + sr) * 512 + (col0g + scc);
        outg[(size_t)gtok * 64 + nt * 16 + lN] = acc[r] + bias;
      }
    }
  }
}

extern "C" void kernel_launch(void* const* d_in, const int* in_sizes, int n_in,
                              void* d_out, int out_size, void* d_ws, size_t ws_size,
                              hipStream_t stream) {
  // inputs: x, H, W, w_qkv, b_qkv, w_pe, b_pe, w_out, b_out  (H=W=512 fixed)
  const float* x    = (const float*)d_in[0];
  const float* wqkv = (const float*)d_in[3];
  const float* bqkv = (const float*)d_in[4];
  const float* wpe  = (const float*)d_in[5];
  const float* bpe  = (const float*)d_in[6];
  const float* wout = (const float*)d_in[7];
  const float* bout = (const float*)d_in[8];
  f16* wsh = (f16*)d_ws;

  const int B = in_sizes[0] / (512 * 512 * 64);
  lml_prep_weights<<<64, 256, 0, stream>>>(wqkv, wout, wsh);
  lml_fused<<<B * 4096, 128, 0, stream>>>(x, wsh, bqkv, wpe, bpe, bout, (float*)d_out);
}
